// RNNSeq2Seq_59897613910164
// MI455X (gfx1250) — matrix operand order repaired and checked by reading, not compiled, not run on hardware
//
#include <hip/hip_runtime.h>

// ---------------------------------------------------------------------------
// RNN seq2seq (2-layer LSTM encoder + 2-layer LSTM decoder + projection)
// CDNA5 / gfx1250: bf16 WMMA (v_wmma_f32_16x16x32_bf16) everywhere,
// persistent grid-synced scan kernel with register-resident cell state.
// Async global->LDS staging (ASYNCcnt path) for the per-step h broadcast.
// ---------------------------------------------------------------------------

#define B_ 16
#define T_ 1024
#define V_ 256
#define E_ 512
#define H_ 1024
#define C_ 256
#define G4H 4096
#define NWG_LSTM 32

#define AS1 __attribute__((address_space(1)))
#define AS3 __attribute__((address_space(3)))

#if defined(__has_builtin)
#if __has_builtin(__builtin_amdgcn_global_load_async_to_lds_b128)
#define HAVE_ASYNC_LDS 1
#endif
#endif

typedef __attribute__((ext_vector_type(16))) __bf16       v16bf;
typedef __attribute__((ext_vector_type(8)))  float        v8f;
typedef __attribute__((ext_vector_type(4)))  unsigned int u32x4;
typedef int v4i_vs __attribute__((vector_size(16)));      // matches builtin param

union FragU { u32x4 u[2]; v16bf v; };

static __device__ __forceinline__ unsigned short f2bf(float f) {
  unsigned u = __float_as_uint(f);
  unsigned r = u + 0x7FFFu + ((u >> 16) & 1u);   // round-nearest-even
  return (unsigned short)(r >> 16);
}
static __device__ __forceinline__ float bf2f(unsigned short h) {
  return __uint_as_float(((unsigned)h) << 16);
}
static __device__ __forceinline__ float sigm(float x) {
  return 1.f / (1.f + __expf(-x));
}
static __device__ __forceinline__ float tanh_f(float x) {
  float e = __expf(-2.f * x);
  return (1.f - e) / (1.f + e);
}

// Load a 16-element bf16 fragment as two b128 loads at p and p+off2 (elements).
static __device__ __forceinline__ v16bf load_frag(const unsigned short* p, int off2) {
  FragU f;
  f.u[0] = *(const u32x4*)p;
  f.u[1] = *(const u32x4*)(p + off2);
  return f.v;
}

static __device__ __forceinline__ v8f wmma_bf16(v16bf a, v16bf b, v8f c) {
  return __builtin_amdgcn_wmma_f32_16x16x32_bf16(false, a, false, b, (short)0, c,
                                                 false, false);
}

// ---------------------------------------------------------------------------
// Utility kernels
// ---------------------------------------------------------------------------
__global__ void k_zero32(unsigned int* __restrict__ p, size_t n) {
  size_t i = (size_t)blockIdx.x * blockDim.x + threadIdx.x;
  size_t s = (size_t)gridDim.x * blockDim.x;
  for (; i < n; i += s) p[i] = 0u;
}

__global__ void k_f32_to_bf16(const float* __restrict__ src,
                              unsigned short* __restrict__ dst, size_t n) {
  size_t i = (size_t)blockIdx.x * blockDim.x + threadIdx.x;
  size_t s = (size_t)gridDim.x * blockDim.x;
  for (; i < n; i += s) dst[i] = f2bf(src[i]);
}

// Embedding gather: out[(t*B+b)*E + e] = bf16(tab[x[b,t]*E + e]). One block per (t,b).
__global__ void k_gather(const int* __restrict__ tok, const float* __restrict__ tab,
                         unsigned short* __restrict__ out, int Edim) {
  int tb = blockIdx.x;          // t*B + b
  int t  = tb >> 4;
  int b  = tb & 15;
  int token = tok[b * T_ + t];
  const float* src = tab + (size_t)token * Edim;
  unsigned short* dst = out + (size_t)tb * Edim;
  for (int e = threadIdx.x; e < Edim; e += blockDim.x) dst[e] = f2bf(src[e]);
}

// ---------------------------------------------------------------------------
// Input-projection GEMM: out(16384 x 4096) = A(16384 x K) @ W(4096 x K)^T + bih+bhh
// bf16 in / f32 accumulate / bf16 out.  Block = 8 waves; wave -> 32x32 macro-tile.
// Grid: (rows/64, 4096/128).
// ---------------------------------------------------------------------------
__global__ __launch_bounds__(256) void k_gemm_bias(
    const unsigned short* __restrict__ A, const unsigned short* __restrict__ W,
    const float* __restrict__ b0, const float* __restrict__ b1,
    unsigned short* __restrict__ out, int K) {
  const int lane = threadIdx.x & 31;
  const int wave = threadIdx.x >> 5;
  const int r0 = blockIdx.x * 64 + (wave >> 2) * 32;
  const int n0 = blockIdx.y * 128 + (wave & 3) * 32;
  const int lan = lane & 15;
  const int kHalfA = (lane & 16) ? 16 : 0;
  const int kHalfB = (lane & 16) ? 16 : 0;

  v8f acc00 = {}, acc01 = {}, acc10 = {}, acc11 = {};
  for (int kc = 0; kc < K; kc += 32) {
    const unsigned short* pa0 = A + (size_t)(r0 + lan) * K + kc + kHalfA;
    const unsigned short* pa1 = A + (size_t)(r0 + 16 + lan) * K + kc + kHalfA;
    const unsigned short* pw0 = W + (size_t)(n0 + lan) * K + kc + kHalfB;
    const unsigned short* pw1 = W + (size_t)(n0 + 16 + lan) * K + kc + kHalfB;
    __builtin_prefetch(pw0 + 32, 0, 1);   // global_prefetch_b8: next K-chunk
    v16bf a0 = load_frag(pa0, 8);
    v16bf a1 = load_frag(pa1, 8);
    v16bf w0 = load_frag(pw0, 8);
    v16bf w1 = load_frag(pw1, 8);
    acc00 = wmma_bf16(a0, w0, acc00);
    acc01 = wmma_bf16(a0, w1, acc01);
    acc10 = wmma_bf16(a1, w0, acc10);
    acc11 = wmma_bf16(a1, w1, acc11);
  }
  const int rOff = (lane & 16) ? 8 : 0;
  v8f* accs[2][2] = {{&acc00, &acc01}, {&acc10, &acc11}};
#pragma unroll
  for (int mt = 0; mt < 2; ++mt) {
#pragma unroll
    for (int nt = 0; nt < 2; ++nt) {
      int n = n0 + nt * 16 + lan;
      float bias = b0[n] + b1[n];
#pragma unroll
      for (int r = 0; r < 8; ++r) {
        int row = r0 + mt * 16 + r + rOff;
        out[(size_t)row * G4H + n] = f2bf((*accs[mt][nt])[r] + bias);
      }
    }
  }
}

// ---------------------------------------------------------------------------
// Persistent LSTM scan.  32 WGs x 256 threads. WG owns 2 j-tiles (32 cols of H).
// Wave layout: cg = wave&1 selects j-tile, ks = wave>>1 selects K slice of 256
// (of K=1024). Partial accumulators reduced via LDS; cell state c lives in
// registers of the ks==0 waves for all T steps. Grid-wide sync per step via
// per-step atomic counters; h broadcast through a double-buffered global array
// staged into LDS each step (async global->LDS when available).
// ---------------------------------------------------------------------------
__global__ __launch_bounds__(256) void k_lstm(
    const unsigned short* __restrict__ xg,   // (T*B, 4H) bf16, bias pre-folded
    const unsigned short* __restrict__ Whh,  // (4H, H) bf16 row-major
    unsigned short* __restrict__ hbuf,       // 2 * (B*H) bf16 double buffer
    unsigned short* __restrict__ ys,         // (T*B, H) bf16
    const float* __restrict__ c0,            // (B, H) f32 initial cell state
    float* __restrict__ hfin, float* __restrict__ cfin,  // (B, H) f32 finals
    int* __restrict__ syncc, int T) {
  __shared__ __align__(16) unsigned short sh_h[B_ * H_];   // 32 KB
  __shared__ float sh_part[3 * 2 * 4 * 8 * 32];            // 24 KB partials

  const int tid  = threadIdx.x;
  const int lane = tid & 31;
  const int wave = tid >> 5;
  const int cg   = wave & 1;   // which of the 2 j-tiles of this WG
  const int ks   = wave >> 1;  // K slice 0..3 (256 K each)
  const int jt   = blockIdx.x * 2 + cg;
  const int j0   = jt * 16;
  const int lan  = lane & 15;
  const int rOff = (lane & 16) ? 8 : 0;
  const int kHalfA = (lane & 16) ? 16 : 0;
  const int kHalfB = (lane & 16) ? 16 : 0;

  v8f cfr = {};
  if (ks == 0) {
#pragma unroll
    for (int r = 0; r < 8; ++r) cfr[r] = c0[(size_t)(r + rOff) * H_ + j0 + lan];
  }

  for (int t = 0; t < T; ++t) {
    // ---- stage h(t) from global double buffer into LDS ([b][k] row-major)
    const unsigned short* hsrc = hbuf + (size_t)(t & 1) * (B_ * H_);
#ifdef HAVE_ASYNC_LDS
    {
#pragma unroll
      for (int i = 0; i < 8; ++i) {
        int e = (i * 256 + tid) * 8;                  // 16B per lane per issue
        __builtin_amdgcn_global_load_async_to_lds_b128(
            (AS1 v4i_vs*)(hsrc + e), (AS3 v4i_vs*)(sh_h + e), 0, 0);
      }
#if __has_builtin(__builtin_amdgcn_s_wait_asynccnt)
      __builtin_amdgcn_s_wait_asynccnt(0);
#else
      asm volatile("s_wait_asynccnt 0x0" ::: "memory");
#endif
    }
#else
#pragma unroll
    for (int i = 0; i < 8; ++i) {
      int e = (i * 256 + tid) * 8;
      *(u32x4*)(sh_h + e) = *(const u32x4*)(hsrc + e);
    }
#endif
    __syncthreads();

    // ---- partial GEMM over this wave's K slice: 4 gate tiles of 16 cols
    v8f acc[4] = {};
    for (int kc = ks * 256; kc < ks * 256 + 256; kc += 32) {
      v16bf a = load_frag(sh_h + lan * H_ + kc + kHalfA, 8);   // ds_load_b128 x2
#pragma unroll
      for (int g = 0; g < 4; ++g) {
        const unsigned short* wp =
            Whh + (size_t)(g * H_ + j0 + lan) * H_ + kc + kHalfB;
        v16bf w = load_frag(wp, 8);
        acc[g] = wmma_bf16(a, w, acc[g]);
      }
    }

    // ---- cross-wave K reduction through LDS
    if (ks > 0) {
      float* pb = sh_part + ((ks - 1) * 2 + cg) * 1024;
#pragma unroll
      for (int g = 0; g < 4; ++g)
#pragma unroll
        for (int r = 0; r < 8; ++r) pb[g * 256 + r * 32 + lane] = acc[g][r];
    }
    __syncthreads();

    if (ks == 0) {
#pragma unroll
      for (int s = 1; s < 4; ++s) {
        const float* pb = sh_part + ((s - 1) * 2 + cg) * 1024;
#pragma unroll
        for (int g = 0; g < 4; ++g)
#pragma unroll
          for (int r = 0; r < 8; ++r) acc[g][r] += pb[g * 256 + r * 32 + lane];
      }
      // ---- gates -> cell update -> h(t+1); c stays in registers
      unsigned short* hdst = hbuf + (size_t)((t + 1) & 1) * (B_ * H_);
#pragma unroll
      for (int r = 0; r < 8; ++r) {
        int b = r + rOff;
        int j = j0 + lan;
        const unsigned short* xr = xg + ((size_t)t * B_ + b) * G4H;
        float gi = acc[0][r] + bf2f(xr[j]);
        float gf = acc[1][r] + bf2f(xr[H_ + j]);
        float gg = acc[2][r] + bf2f(xr[2 * H_ + j]);
        float go = acc[3][r] + bf2f(xr[3 * H_ + j]);
        float iv = sigm(gi), fv = sigm(gf), gv = tanh_f(gg), ov = sigm(go);
        float c = fv * cfr[r] + iv * gv;
        cfr[r] = c;
        float h = ov * tanh_f(c);
        unsigned short hb = f2bf(h);
        ys[((size_t)t * B_ + b) * H_ + j] = hb;
        hdst[(size_t)b * H_ + j] = hb;
        if (t == T - 1) {
          hfin[(size_t)b * H_ + j] = h;
          cfin[(size_t)b * H_ + j] = c;
        }
      }
    }

    // ---- grid-wide step barrier (per-step counter, release/acquire)
    __threadfence();
    __syncthreads();
    if (tid == 0) {
      __hip_atomic_fetch_add(&syncc[t], 1, __ATOMIC_ACQ_REL,
                             __HIP_MEMORY_SCOPE_AGENT);
      while (__hip_atomic_load(&syncc[t], __ATOMIC_ACQUIRE,
                               __HIP_MEMORY_SCOPE_AGENT) < NWG_LSTM)
        __builtin_amdgcn_s_sleep(2);
    }
    __syncthreads();
  }
}

// ---------------------------------------------------------------------------
// Output projection: logits(B,C,T) = Yd(t*B+b, H) @ Wo(C,H)^T + bo.
// Wave: one t (row-tile = all 16 batches) x 64 output cols. Grid: T/2 blocks.
// ---------------------------------------------------------------------------
__global__ __launch_bounds__(256) void k_outproj(
    const unsigned short* __restrict__ Yd, const unsigned short* __restrict__ Wo,
    const float* __restrict__ bo, float* __restrict__ out) {
  const int lane = threadIdx.x & 31;
  const int wave = threadIdx.x >> 5;
  const int t  = blockIdx.x * 2 + (wave >> 2);
  const int c0 = (wave & 3) * 64;
  const int lan = lane & 15;
  const int rOff = (lane & 16) ? 8 : 0;
  const int kHalfA = (lane & 16) ? 16 : 0;
  const int kHalfB = (lane & 16) ? 16 : 0;

  v8f acc[4] = {};
  for (int kc = 0; kc < H_; kc += 32) {
    v16bf a = load_frag(Yd + ((size_t)t * B_ + lan) * H_ + kc + kHalfA, 8);
#pragma unroll
    for (int ct = 0; ct < 4; ++ct) {
      v16bf w = load_frag(Wo + (size_t)(c0 + ct * 16 + lan) * H_ + kc + kHalfB, 8);
      acc[ct] = wmma_bf16(a, w, acc[ct]);
    }
  }
#pragma unroll
  for (int ct = 0; ct < 4; ++ct) {
    int c = c0 + ct * 16 + lan;
    float bias = bo[c];
#pragma unroll
    for (int r = 0; r < 8; ++r) {
      int b = r + rOff;
      out[((size_t)b * C_ + c) * T_ + t] = acc[ct][r] + bias;
    }
  }
}

// ---------------------------------------------------------------------------
// Host-side orchestration
// ---------------------------------------------------------------------------
static inline size_t aln(size_t x) { return (x + 255) & ~(size_t)255; }

extern "C" void kernel_launch(void* const* d_in, const int* in_sizes, int n_in,
                              void* d_out, int out_size, void* d_ws, size_t ws_size,
                              hipStream_t stream) {
  (void)in_sizes; (void)n_in; (void)out_size; (void)ws_size;
  const int*   x       = (const int*)d_in[0];
  const float* enc_emb = (const float*)d_in[1];
  const float* dec_emb = (const float*)d_in[2];
  const float* Wih[4]  = {(const float*)d_in[3],  (const float*)d_in[7],
                          (const float*)d_in[11], (const float*)d_in[15]};
  const float* Whh[4]  = {(const float*)d_in[4],  (const float*)d_in[8],
                          (const float*)d_in[12], (const float*)d_in[16]};
  const float* bih[4]  = {(const float*)d_in[5],  (const float*)d_in[9],
                          (const float*)d_in[13], (const float*)d_in[17]};
  const float* bhh[4]  = {(const float*)d_in[6],  (const float*)d_in[10],
                          (const float*)d_in[14], (const float*)d_in[18]};
  const float* out_W = (const float*)d_in[19];
  const float* out_b = (const float*)d_in[20];
  float* logits = (float*)d_out;

  // ---- carve workspace
  char* w = (char*)d_ws;
  auto carve = [&](size_t bytes) -> void* { void* p = w; w += aln(bytes); return p; };
  int*            syncc = (int*)carve(T_ * sizeof(int));
  unsigned short* hbuf  = (unsigned short*)carve(2 * B_ * H_ * 2);
  float*          zeros = (float*)carve(B_ * H_ * 4);
  float *hf[4], *cf[4];
  for (int l = 0; l < 4; ++l) { hf[l] = (float*)carve(B_ * H_ * 4);
                                cf[l] = (float*)carve(B_ * H_ * 4); }
  const size_t wihN[4] = {(size_t)G4H * E_, (size_t)G4H * H_,
                          (size_t)G4H * E_, (size_t)G4H * H_};
  unsigned short *WihB[4], *WhhB[4];
  for (int l = 0; l < 4; ++l) WihB[l] = (unsigned short*)carve(wihN[l] * 2);
  for (int l = 0; l < 4; ++l) WhhB[l] = (unsigned short*)carve((size_t)G4H * H_ * 2);
  unsigned short* WoB = (unsigned short*)carve((size_t)C_ * H_ * 2);
  unsigned short* emb = (unsigned short*)carve((size_t)B_ * T_ * E_ * 2);
  unsigned short* xg  = (unsigned short*)carve((size_t)B_ * T_ * G4H * 2);
  unsigned short* ysA = (unsigned short*)carve((size_t)B_ * T_ * H_ * 2);
  unsigned short* ysB = (unsigned short*)carve((size_t)B_ * T_ * H_ * 2);

  // ---- one-time per call: zero init + weight bf16 conversion (all in L2)
  k_zero32<<<64, 256, 0, stream>>>((unsigned int*)zeros, (size_t)B_ * H_);
  for (int l = 0; l < 4; ++l) {
    k_f32_to_bf16<<<512, 256, 0, stream>>>(Wih[l], WihB[l], wihN[l]);
    k_f32_to_bf16<<<512, 256, 0, stream>>>(Whh[l], WhhB[l], (size_t)G4H * H_);
  }
  k_f32_to_bf16<<<256, 256, 0, stream>>>(out_W, WoB, (size_t)C_ * H_);

  const dim3 gE(T_ * B_ / 64, G4H / 128);  // K = E layers
  const dim3 gH(T_ * B_ / 64, G4H / 128);  // K = H layers

  // ======== encoder layer 0 ========
  k_gather<<<T_ * B_, 256, 0, stream>>>(x, enc_emb, emb, E_);
  k_gemm_bias<<<gE, 256, 0, stream>>>(emb, WihB[0], bih[0], bhh[0], xg, E_);
  k_zero32<<<8, 256, 0, stream>>>((unsigned int*)syncc, T_);
  k_zero32<<<16, 256, 0, stream>>>((unsigned int*)hbuf, (size_t)B_ * H_);  // h0=0 bf16
  k_lstm<<<NWG_LSTM, 256, 0, stream>>>(xg, WhhB[0], hbuf, ysA, zeros,
                                       hf[0], cf[0], syncc, T_);
  // ======== encoder layer 1 ========
  k_gemm_bias<<<gH, 256, 0, stream>>>(ysA, WihB[1], bih[1], bhh[1], xg, H_);
  k_zero32<<<8, 256, 0, stream>>>((unsigned int*)syncc, T_);
  k_zero32<<<16, 256, 0, stream>>>((unsigned int*)hbuf, (size_t)B_ * H_);
  k_lstm<<<NWG_LSTM, 256, 0, stream>>>(xg, WhhB[1], hbuf, ysB, zeros,
                                       hf[1], cf[1], syncc, T_);
  // ======== decoder layer 0 (init from encoder layer-0 finals) ========
  k_gather<<<T_ * B_, 256, 0, stream>>>(x, dec_emb, emb, E_);
  k_gemm_bias<<<gE, 256, 0, stream>>>(emb, WihB[2], bih[2], bhh[2], xg, E_);
  k_zero32<<<8, 256, 0, stream>>>((unsigned int*)syncc, T_);
  k_f32_to_bf16<<<32, 256, 0, stream>>>(hf[0], hbuf, (size_t)B_ * H_);
  k_lstm<<<NWG_LSTM, 256, 0, stream>>>(xg, WhhB[2], hbuf, ysA, cf[0],
                                       hf[2], cf[2], syncc, T_);
  // ======== decoder layer 1 (init from encoder layer-1 finals) ========
  k_gemm_bias<<<gH, 256, 0, stream>>>(ysA, WihB[3], bih[3], bhh[3], xg, H_);
  k_zero32<<<8, 256, 0, stream>>>((unsigned int*)syncc, T_);
  k_f32_to_bf16<<<32, 256, 0, stream>>>(hf[1], hbuf, (size_t)B_ * H_);
  k_lstm<<<NWG_LSTM, 256, 0, stream>>>(xg, WhhB[3], hbuf, ysB, cf[1],
                                       hf[3], cf[3], syncc, T_);
  // ======== output projection ========
  k_outproj<<<T_ / 2, 256, 0, stream>>>(ysB, WoB, out_b, logits);
}
